// CSNN_6906307412656
// MI455X (gfx1250) — matrix operand order repaired and checked by reading, not compiled, not run on hardware
//
#include <hip/hip_runtime.h>
#include <hip/hip_bf16.h>

typedef __attribute__((ext_vector_type(16))) _Float16 v16h;
typedef __attribute__((ext_vector_type(8)))  _Float16 v8h;
typedef __attribute__((ext_vector_type(8)))  float    v8f;
typedef __attribute__((ext_vector_type(4)))  int      v4i;

#define T_STEPS 100
#define EPSBN 1e-5f

#define SHUF16(lo, hi) __builtin_shufflevector(lo, hi, 0,1,2,3,4,5,6,7,8,9,10,11,12,13,14,15)

#if __has_builtin(__builtin_amdgcn_global_load_async_to_lds_b128) && \
    __has_builtin(__builtin_amdgcn_s_wait_asynccnt)
#define USE_ASYNC_LDS 1
#define GPTR(p) ((__attribute__((address_space(1))) v4i*)(p))
#define LPTR(p) ((__attribute__((address_space(3))) v4i*)(p))
#else
#define USE_ASYNC_LDS 0
#endif

// ---------------------------------------------------------------------------
// conv1 (1->64, 3x3 SAME, 28x28) + BN1 folded. One output per thread.
// h1 layout: [B=16][C=64][28][28] fp32
// ---------------------------------------------------------------------------
__global__ void k_conv1_bn(const float* __restrict__ x, const float* __restrict__ W1,
                           const float* __restrict__ g1, const float* __restrict__ b1,
                           const float* __restrict__ m1, const float* __restrict__ v1,
                           float* __restrict__ h1) {
    int tid = blockIdx.x * blockDim.x + threadIdx.x;          // 802816
    int xx = tid % 28, yy = (tid / 28) % 28;
    int c  = (tid / 784) % 64, b = tid / (784 * 64);
    const float* xb = x + b * 784;
    const float* w  = W1 + c * 9;
    float s = 0.f;
#pragma unroll
    for (int ky = 0; ky < 3; ++ky) {
        int sy = yy + ky - 1;
        if ((unsigned)sy >= 28u) continue;
#pragma unroll
        for (int kx = 0; kx < 3; ++kx) {
            int sx = xx + kx - 1;
            if ((unsigned)sx >= 28u) continue;
            s += xb[sy * 28 + sx] * w[ky * 3 + kx];
        }
    }
    float inv = g1[c] * rsqrtf(v1[c] + EPSBN);
    h1[tid] = s * inv + (b1[c] - m1[c] * inv);
}

// ---------------------------------------------------------------------------
// LIF layer 1 + 2x2 maxpool (OR of 4 binary spike trains), fused.
// p1 layout (NHWC for implicit GEMM): [t*16+b][14][14][64] f16 (values 0/1)
// ---------------------------------------------------------------------------
__global__ void k_lif1_pool(const float* __restrict__ h1, _Float16* __restrict__ p1) {
    int tid = blockIdx.x * blockDim.x + threadIdx.x;          // 200704
    int c = tid & 63;
    int rest = tid >> 6;
    int px = rest % 14, py = (rest / 14) % 14, b = rest / 196;
    int ib = ((b * 64 + c) * 28 + 2 * py) * 28 + 2 * px;
    float h00 = h1[ib], h01 = h1[ib + 1], h10 = h1[ib + 28], h11 = h1[ib + 29];
    float u0 = 0.f, u1 = 0.f, u2 = 0.f, u3 = 0.f;
    int ob = (b * 196 + py * 14 + px) * 64 + c;
    for (int t = 0; t < T_STEPS; ++t) {
        u0 = 0.5f * (u0 + h00); u1 = 0.5f * (u1 + h01);
        u2 = 0.5f * (u2 + h10); u3 = 0.5f * (u3 + h11);
        float s0 = (u0 >= 1.f) ? 1.f : 0.f; u0 -= u0 * s0;
        float s1 = (u1 >= 1.f) ? 1.f : 0.f; u1 -= u1 * s1;
        float s2 = (u2 >= 1.f) ? 1.f : 0.f; u2 -= u2 * s2;
        float s3 = (u3 >= 1.f) ? 1.f : 0.f; u3 -= u3 * s3;
        float sp = fmaxf(fmaxf(s0, s1), fmaxf(s2, s3));
        p1[ob + t * 200704] = (_Float16)sp;
    }
}

// ---------------------------------------------------------------------------
// Pre-swizzle conv2 weights into per-lane WMMA B-fragment layout, BN2 scale
// folded. k = tap*64 + cin.  Bp2[((kc*64 + n)*2 + kg)*16 + j].
// ---------------------------------------------------------------------------
__global__ void k_prep_w2(const float* __restrict__ W2, const float* __restrict__ g2,
                          const float* __restrict__ v2, _Float16* __restrict__ Bp2) {
    int e = blockIdx.x * blockDim.x + threadIdx.x;            // 36864
    int j = e % 16, kg = (e / 16) & 1, n = (e / 32) % 64, kc = e / (32 * 64);
    int k = kc * 32 + kg * 16 + j;
    int tap = k / 64, cin = k % 64;
    float inv = g2[n] * rsqrtf(v2[n] + EPSBN);
    Bp2[e] = (_Float16)(W2[(n * 64 + cin) * 9 + tap] * inv);
}

// Pre-swizzle FC1 weights.  Bp[((kc*1024 + n)*2 + kg)*16 + j], 98*1024*32.
__global__ void k_prep_wfc1(const float* __restrict__ Wfc1, _Float16* __restrict__ Bp) {
    int e = blockIdx.x * blockDim.x + threadIdx.x;            // 3211264
    int j = e % 16, kg = (e / 16) & 1, n = (e / 32) % 1024, kc = e / (32 * 1024);
    int k = kc * 32 + kg * 16 + j;
    Bp[e] = (_Float16)Wfc1[n * 3136 + k];
}

// ---------------------------------------------------------------------------
// conv2 as implicit GEMM via WMMA f32_16x16x32_f16.
// M = 313600 pixels (256/block), N = 64, K = 576 (18 chunks of 32).
// Full pre-swizzled B (72KB) resident in LDS; A patch tile staged via LDS
// with register prefetch of the next K-chunk.  Each wave: 32(M) x 64(N)
// -> 2 A fragments, 8 WMMAs per K-step (B fragments amortized over 2x math).
// h2: [t*16+b][64][196] fp32 (BN2 beta added in LIF2).
// ---------------------------------------------------------------------------
__global__ __launch_bounds__(256)
void k_conv2_gemm(const _Float16* __restrict__ p1, const _Float16* __restrict__ Bp2,
                  float* __restrict__ h2) {
    __shared__ __align__(16) _Float16 Bs[18 * 2048];          // 73728 B
    __shared__ __align__(16) _Float16 As[256][40];            // 20480 B
    const int tid = threadIdx.x;
    const int lane = tid & 31, wv = tid >> 5;
    const int rowBase = blockIdx.x * 256;                     // 1225 blocks

    // ---- preload all of B into LDS (coalesced, 16B per thread per step)
#pragma unroll
    for (int j = 0; j < 18; ++j) {
        int idx = (j * 256 + tid) * 8;
        *(v8h*)&Bs[idx] = *(const v8h*)(Bp2 + idx);
    }

    v8f acc[2][4];
    const v8f vzero = {0.f, 0.f, 0.f, 0.f, 0.f, 0.f, 0.f, 0.f};
#pragma unroll
    for (int mt = 0; mt < 2; ++mt)
#pragma unroll
        for (int nt = 0; nt < 4; ++nt) acc[mt][nt] = vzero;

    // staging role: one thread per A row (32 halves = 64B)
    const int sr = tid;
    const int sp = rowBase + sr;
    const int stb = sp / 196, spix = sp % 196;
    const int spy = spix / 14, spx = spix % 14;

    // fragment role: wave covers rows wv*32 .. wv*32+31
    const int arow = wv * 32 + (lane & 15);
    const int kg = lane >> 4;
    const int col = lane & 15;

    const v8h vh0 = (v8h)(_Float16)0.0f;
    auto loadA = [&](int kc, v8h* d) {
        const int tap = kc >> 1, c0 = (kc & 1) * 32;
        const int sy = spy + tap / 3 - 1, sx = spx + tap % 3 - 1;
        d[0] = vh0; d[1] = vh0; d[2] = vh0; d[3] = vh0;
        if ((unsigned)sy < 14u && (unsigned)sx < 14u) {
            const _Float16* src = p1 + ((stb * 196 + sy * 14 + sx) * 64 + c0);
            d[0] = *(const v8h*)src;
            d[1] = *(const v8h*)(src + 8);
            d[2] = *(const v8h*)(src + 16);
            d[3] = *(const v8h*)(src + 24);
        }
    };

    v8h pa[4];
    loadA(0, pa);

    for (int kc = 0; kc < 18; ++kc) {
        __syncthreads();                      // prior LDS reads done (also covers Bs fill)
        *(v8h*)&As[sr][0]  = pa[0];
        *(v8h*)&As[sr][8]  = pa[1];
        *(v8h*)&As[sr][16] = pa[2];
        *(v8h*)&As[sr][24] = pa[3];
        __syncthreads();
        if (kc + 1 < 18) loadA(kc + 1, pa);   // overlap with compute below

        // A fragments in contiguous k order (k = 16*kg + s), matching Bp2 (two M subtiles)
        v16h a[2];
#pragma unroll
        for (int mt = 0; mt < 2; ++mt) {
            v8h al = *(const v8h*)&As[arow + mt * 16][kg * 16];
            v8h ah = *(const v8h*)&As[arow + mt * 16][kg * 16 + 8];
            a[mt] = SHUF16(al, ah);
        }

        // all B fragments from LDS, hoisted ahead of the WMMA chain
        const _Float16* bb = &Bs[kc * 2048];
        v16h bv[4];
#pragma unroll
        for (int nt = 0; nt < 4; ++nt) {
            const _Float16* bl = bb + (((nt * 16 + col) * 2 + kg) * 16);
            v8h blo = *(const v8h*)bl;
            v8h bhi = *(const v8h*)(bl + 8);
            bv[nt] = SHUF16(blo, bhi);
        }
#pragma unroll
        for (int mt = 0; mt < 2; ++mt)
#pragma unroll
            for (int nt = 0; nt < 4; ++nt)
                acc[mt][nt] = __builtin_amdgcn_wmma_f32_16x16x32_f16(
                    false, a[mt], false, bv[nt], (short)0, acc[mt][nt], false, false);
    }

    // store per 32-bit C/D layout: vgpr i -> M = i + 8*(lane/16), N = lane%16
#pragma unroll
    for (int mt = 0; mt < 2; ++mt)
#pragma unroll
    for (int nt = 0; nt < 4; ++nt) {
        int n = nt * 16 + col;
#pragma unroll
        for (int i = 0; i < 8; ++i) {
            int pr = rowBase + wv * 32 + mt * 16 + i + 8 * kg;
            int tb2 = pr / 196, pix2 = pr % 196;
            h2[(tb2 * 64 + n) * 196 + pix2] = acc[mt][nt][i];
        }
    }
}

// ---------------------------------------------------------------------------
// LIF layer 2 + 2x2 maxpool fused; BN2 beta added here; T-scan in registers.
// p2 layout: [t*16+b][c*49 + 7x7pix] f16
// ---------------------------------------------------------------------------
__global__ void k_lif2_pool(const float* __restrict__ h2,
                            const float* __restrict__ g2, const float* __restrict__ b2,
                            const float* __restrict__ m2, const float* __restrict__ v2,
                            _Float16* __restrict__ p2) {
    int tid = blockIdx.x * blockDim.x + threadIdx.x;          // 50176
    int pp = tid % 49;
    int c  = (tid / 49) % 64;
    int b  = tid / (49 * 64);
    int py = pp / 7, px = pp % 7;
    float inv  = g2[c] * rsqrtf(v2[c] + EPSBN);
    float beta = b2[c] - m2[c] * inv;
    int base = (b * 64 + c) * 196 + (2 * py) * 14 + 2 * px;   // stride_t = 200704
    int ob = b * 3136 + c * 49 + pp;                          // stride_t = 16*3136
    float u0 = 0.f, u1 = 0.f, u2 = 0.f, u3 = 0.f;
    for (int t = 0; t < T_STEPS; ++t) {
        const float* ht = h2 + t * 200704 + base;
        float x0 = ht[0] + beta, x1 = ht[1] + beta, x2 = ht[14] + beta, x3 = ht[15] + beta;
        u0 = 0.5f * (u0 + x0); u1 = 0.5f * (u1 + x1);
        u2 = 0.5f * (u2 + x2); u3 = 0.5f * (u3 + x3);
        float s0 = (u0 >= 1.f) ? 1.f : 0.f; u0 -= u0 * s0;
        float s1 = (u1 >= 1.f) ? 1.f : 0.f; u1 -= u1 * s1;
        float s2 = (u2 >= 1.f) ? 1.f : 0.f; u2 -= u2 * s2;
        float s3 = (u3 >= 1.f) ? 1.f : 0.f; u3 -= u3 * s3;
        p2[ob + t * (16 * 3136)] = (_Float16)fmaxf(fmaxf(s0, s1), fmaxf(s2, s3));
    }
}

// ---------------------------------------------------------------------------
// FC1 GEMM via WMMA: M=1600, K=3136, N=1024.
// Block = 64 rows x 256 cols; waves 2(M) x 4(N); each wave 32x64
// -> 2 A fragments, 8 WMMAs per K-step.
// A tile staged with CDNA5 async global->LDS copies (double-buffered,
// ASYNCcnt-tracked) when available; B fragment loads hoisted per K-step.
// ---------------------------------------------------------------------------
__global__ __launch_bounds__(256)
void k_fc1_gemm(const _Float16* __restrict__ p2, const _Float16* __restrict__ Bp,
                float* __restrict__ h3) {
#if USE_ASYNC_LDS
    __shared__ __align__(16) _Float16 As[2][64][40];
#else
    __shared__ __align__(16) _Float16 As1[64][40];
#endif
    const int tid = threadIdx.x;
    const int lane = tid & 31, wv = tid >> 5;
    const int rowBase = blockIdx.x * 64;                      // 25 blocks in x
    const int colBase = blockIdx.y * 256;                     // 4 blocks in y
    const int mr = (wv & 1) * 32;
    const int nc = colBase + (wv >> 1) * 64;

    v8f acc[2][4];
    const v8f vzero = {0.f, 0.f, 0.f, 0.f, 0.f, 0.f, 0.f, 0.f};
#pragma unroll
    for (int mt = 0; mt < 2; ++mt)
#pragma unroll
        for (int nt = 0; nt < 4; ++nt) acc[mt][nt] = vzero;

    const int sr = tid >> 2;            // 64 rows
    const int ko = (tid & 3) * 8;       // 8 halves each (16B = one b128)
    const _Float16* srcBase = p2 + (rowBase + sr) * 3136 + ko;
    const int arow = mr + (lane & 15);
    const int kg = lane >> 4;
    const int col = lane & 15;

#if USE_ASYNC_LDS
    __builtin_amdgcn_global_load_async_to_lds_b128(
        GPTR(srcBase), LPTR(&As[0][sr][ko]), 0, 0);
#else
    v8h pa = *(const v8h*)srcBase;
#endif

    for (int kc = 0; kc < 98; ++kc) {
#if USE_ASYNC_LDS
        const int cb = kc & 1;
        __builtin_amdgcn_s_wait_asynccnt(0);
        __syncthreads();
        if (kc + 1 < 98)
            __builtin_amdgcn_global_load_async_to_lds_b128(
                GPTR(srcBase + (kc + 1) * 32), LPTR(&As[cb ^ 1][sr][ko]), 0, 0);
        const _Float16 (*Ac)[40] = As[cb];
#else
        __syncthreads();
        *(v8h*)&As1[sr][ko] = pa;
        __syncthreads();
        if (kc + 1 < 98) pa = *(const v8h*)(srcBase + (kc + 1) * 32);
        const _Float16 (*Ac)[40] = As1;
#endif

        // hoist all B fragment loads (L2-resident pre-swizzled weights)
        const _Float16* bb = Bp + kc * 1024 * 32;
        v16h bv[4];
#pragma unroll
        for (int nt = 0; nt < 4; ++nt) {
            const _Float16* bl = bb + (((nc + nt * 16 + col) * 2 + kg) * 16);
            v8h blo = *(const v8h*)bl;
            v8h bhi = *(const v8h*)(bl + 8);
            bv[nt] = SHUF16(blo, bhi);
        }

        v16h a[2];
#pragma unroll
        for (int mt = 0; mt < 2; ++mt) {
            v8h al = *(const v8h*)&Ac[arow + mt * 16][kg * 16];
            v8h ah = *(const v8h*)&Ac[arow + mt * 16][kg * 16 + 8];
            a[mt] = SHUF16(al, ah);
        }

#pragma unroll
        for (int mt = 0; mt < 2; ++mt)
#pragma unroll
            for (int nt = 0; nt < 4; ++nt)
                acc[mt][nt] = __builtin_amdgcn_wmma_f32_16x16x32_f16(
                    false, a[mt], false, bv[nt], (short)0, acc[mt][nt], false, false);
    }

#pragma unroll
    for (int mt = 0; mt < 2; ++mt)
#pragma unroll
    for (int nt = 0; nt < 4; ++nt) {
        int n = nc + nt * 16 + col;
#pragma unroll
        for (int i = 0; i < 8; ++i) {
            int r = rowBase + mr + mt * 16 + i + 8 * kg;
            h3[r * 1024 + n] = acc[mt][nt][i];
        }
    }
}

// LIF layer 3: per-neuron T-scan. s3: [t*16+b][1024] f16
__global__ void k_lif3(const float* __restrict__ h3, _Float16* __restrict__ s3) {
    int tid = blockIdx.x * blockDim.x + threadIdx.x;          // 16384
    int n = tid % 1024, b = tid / 1024;
    float u = 0.f;
    for (int t = 0; t < T_STEPS; ++t) {
        int idx = (t * 16 + b) * 1024 + n;
        u = 0.5f * (u + h3[idx]);
        float s = (u >= 1.f) ? 1.f : 0.f;
        u -= u * s;
        s3[idx] = (_Float16)s;
    }
}

// FC2: one wave per output scalar (m in [0,1600), o in [0,10)); K=1024.
__global__ __launch_bounds__(256)
void k_fc2(const _Float16* __restrict__ s3, const float* __restrict__ Wfc2,
           float* __restrict__ h4) {
    int gw = blockIdx.x * 8 + (threadIdx.x >> 5);             // 16000 waves
    int lane = threadIdx.x & 31;
    int m = gw / 10, o = gw % 10;
    const _Float16* a = s3 + m * 1024;
    const float* w = Wfc2 + o * 1024;
    float part = 0.f;
#pragma unroll 4
    for (int j = lane; j < 1024; j += 32)
        part += (float)a[j] * w[j];
#pragma unroll
    for (int off = 16; off >= 1; off >>= 1)
        part += __shfl_xor(part, off, 32);
    if (lane == 0) h4[m * 10 + o] = part;
}

// LIF layer 4 + mean over T.  out: [B=16][10] fp32
__global__ void k_lif4_mean(const float* __restrict__ h4, float* __restrict__ out) {
    int tid = blockIdx.x * blockDim.x + threadIdx.x;
    if (tid >= 160) return;
    int o = tid % 10, b = tid / 10;
    float u = 0.f, accm = 0.f;
    for (int t = 0; t < T_STEPS; ++t) {
        u = 0.5f * (u + h4[(t * 16 + b) * 10 + o]);
        float s = (u >= 1.f) ? 1.f : 0.f;
        u -= u * s;
        accm += s;
    }
    out[b * 10 + o] = accm * (1.f / (float)T_STEPS);
}

// ---------------------------------------------------------------------------
static inline char* wsAlign(char*& cur, size_t bytes) {
    char* p = cur;
    cur += (bytes + 255) & ~(size_t)255;
    return p;
}

extern "C" void kernel_launch(void* const* d_in, const int* in_sizes, int n_in,
                              void* d_out, int out_size, void* d_ws, size_t ws_size,
                              hipStream_t stream) {
    const float* x    = (const float*)d_in[0];
    const float* W1   = (const float*)d_in[1];
    const float* g1   = (const float*)d_in[2];
    const float* b1   = (const float*)d_in[3];
    const float* m1   = (const float*)d_in[4];
    const float* v1   = (const float*)d_in[5];
    const float* W2   = (const float*)d_in[6];
    const float* g2   = (const float*)d_in[7];
    const float* b2   = (const float*)d_in[8];
    const float* m2   = (const float*)d_in[9];
    const float* v2   = (const float*)d_in[10];
    const float* Wfc1 = (const float*)d_in[11];
    const float* Wfc2 = (const float*)d_in[12];
    float* out = (float*)d_out;

    char* cur = (char*)d_ws;
    float*    h1   = (float*)   wsAlign(cur, (size_t)802816 * 4);
    _Float16* p1   = (_Float16*)wsAlign(cur, (size_t)20070400 * 2);
    float*    h2   = (float*)   wsAlign(cur, (size_t)20070400 * 4);
    _Float16* p2   = (_Float16*)wsAlign(cur, (size_t)5017600 * 2);
    float*    h3   = (float*)   wsAlign(cur, (size_t)1638400 * 4);
    _Float16* s3   = (_Float16*)wsAlign(cur, (size_t)1638400 * 2);
    float*    h4   = (float*)   wsAlign(cur, (size_t)16000 * 4);
    _Float16* Bp2  = (_Float16*)wsAlign(cur, (size_t)36864 * 2);
    _Float16* Bpf1 = (_Float16*)wsAlign(cur, (size_t)3211264 * 2);

    k_conv1_bn  <<<3136, 256, 0, stream>>>(x, W1, g1, b1, m1, v1, h1);
    k_prep_w2   <<<144, 256, 0, stream>>>(W2, g2, v2, Bp2);
    k_prep_wfc1 <<<12544, 256, 0, stream>>>(Wfc1, Bpf1);
    k_lif1_pool <<<784, 256, 0, stream>>>(h1, p1);
    k_conv2_gemm<<<1225, 256, 0, stream>>>(p1, Bp2, h2);
    k_lif2_pool <<<196, 256, 0, stream>>>(h2, g2, b2, m2, v2, p2);
    k_fc1_gemm  <<<dim3(25, 4), 256, 0, stream>>>(p2, Bpf1, h3);
    k_lif3      <<<64, 256, 0, stream>>>(h3, s3);
    k_fc2       <<<2000, 256, 0, stream>>>(s3, Wfc2, h4);
    k_lif4_mean <<<1, 256, 0, stream>>>(h4, out);
}
